// GGAT2Layer_42984032698820
// MI455X (gfx1250) — compile-verified, matrix operand order checked by reading, not run on hardware
//
#include <hip/hip_runtime.h>

typedef __attribute__((ext_vector_type(16))) _Float16 v16h;
typedef __attribute__((ext_vector_type(8)))  float    v8f;

#define NB    64
#define NPG   2048
#define NTOT  (NB * NPG)      // 131072 nodes
#define EPG   32768
#define ETOT  (NB * EPG)      // 2097152 edges
#define DF    32
#define KKEEP (NPG / 2)       // 1024 kept per graph
#define NKEEP (NB * KKEEP)    // 65536 kept total

// ---------------------------------------------------------------- fills
__global__ void fill_f32(float* __restrict__ p, float v, long long n) {
  long long i = (long long)blockIdx.x * blockDim.x + threadIdx.x;
  if (i < n) p[i] = v;
}
__global__ void fill_i32(int* __restrict__ p, int v, int n) {
  int i = blockIdx.x * blockDim.x + threadIdx.x;
  if (i < n) p[i] = v;
}

// ------------------------------------------- weight pre-shuffle to WMMA B layout
// B-fragment layout (mirrors ISA 16-bit operand layout): lane (0..31) owns
// column N = nt*16 + (lane&15); halves j=0..7 hold K = hs*8 + j, j=8..15 hold
// K = 16 + hs*8 + (j-8), where hs = lane>>4.  Stored contiguously: 32B/lane.
__global__ void prep_weights(const float* __restrict__ Wr1, const float* __restrict__ Wl1,
                             const float* __restrict__ Wr2, const float* __restrict__ Wl2,
                             const float* __restrict__ Wr3, const float* __restrict__ Wl3,
                             _Float16* __restrict__ wbuf) {
  int t = threadIdx.x;
  if (t >= 6 * 2 * 32) return;
  int lane = t & 31;
  int nt   = (t >> 5) & 1;
  int mat  = t >> 6;              // 0..5 : Wr1,Wl1,Wr2,Wl2,Wr3,Wl3
  const float* W = (mat == 0) ? Wr1 : (mat == 1) ? Wl1 : (mat == 2) ? Wr2
                 : (mat == 3) ? Wl2 : (mat == 4) ? Wr3 : Wl3;
  int col = nt * 16 + (lane & 15);
  int hs  = lane >> 4;
  _Float16* d = wbuf + (((mat * 2) + nt) * 32 + lane) * 16;
  for (int j = 0; j < 8; ++j) {
    d[j]     = (_Float16)W[(hs * 8 + j) * DF + col];
    d[8 + j] = (_Float16)W[(16 + hs * 8 + j) * DF + col];
  }
}

// ---------------------------------------------------------------- scatter add
// One wave32 per edge: lane d handles dword d of the 32-float row.
// agg (16.7MB) is L2-resident -> atomics resolve in GL2.
__global__ void edge_scatter(const float* __restrict__ x,
                             const int* __restrict__ src, const int* __restrict__ dst,
                             float* __restrict__ agg) {
  long long idx = (long long)blockIdx.x * blockDim.x + threadIdx.x;
  int e = (int)(idx >> 5);
  int d = (int)(idx & 31);
  if (e >= ETOT) return;
  __builtin_prefetch(src + e + 8192, 0, 0);   // global_prefetch_b8 (speculative)
  __builtin_prefetch(dst + e + 8192, 0, 0);
  int s  = src[e];
  int dd = dst[e];
  atomicAdd(agg + (long long)dd * DF + d, x[(long long)s * DF + d]);
}

// ---------------------------------------------------------------- WMMA GEMM + score
__device__ inline v16h load_bfrag(const _Float16* __restrict__ wbuf, int mat, int nt, int lane) {
  return *(const v16h*)(wbuf + (((mat * 2) + nt) * 32 + lane) * 16);
}

__global__ void __launch_bounds__(256) gemm_score(
    const float* __restrict__ x, const float* __restrict__ agg,
    const _Float16* __restrict__ wbuf,
    const float* __restrict__ b1, const float* __restrict__ b2, const float* __restrict__ b3,
    float* __restrict__ x3, float* __restrict__ score) {
  int lane    = threadIdx.x & 31;
  int wv      = threadIdx.x >> 5;
  int tile    = blockIdx.x * 8 + wv;
  int rowbase = tile * 16;
  int m  = lane & 15;
  int hs = lane >> 4;

  // A-fragments (ISA 16-bit A layout): lanes 0-15 hold K {0..7,16..23} of row m,
  // lanes 16-31 hold K {8..15,24..31}.
  const float* xr = x   + (long long)(rowbase + m) * DF;
  const float* ar = agg + (long long)(rowbase + m) * DF;
  int k0 = hs * 8, k1 = 16 + hs * 8;
  v16h a_x, a_g;
#pragma unroll
  for (int j = 0; j < 8; ++j) {
    a_x[j]     = (_Float16)xr[k0 + j];
    a_x[8 + j] = (_Float16)xr[k1 + j];
    a_g[j]     = (_Float16)ar[k0 + j];
    a_g[8 + j] = (_Float16)ar[k1 + j];
  }

  float sg[8];
#pragma unroll
  for (int g = 0; g < 8; ++g) sg[g] = 0.0f;

#pragma unroll
  for (int nt = 0; nt < 2; ++nt) {
    int col = nt * 16 + m;
    v16h br1 = load_bfrag(wbuf, 0, nt, lane), bl1 = load_bfrag(wbuf, 1, nt, lane);
    v16h br2 = load_bfrag(wbuf, 2, nt, lane), bl2 = load_bfrag(wbuf, 3, nt, lane);
    v16h br3 = load_bfrag(wbuf, 4, nt, lane), bl3 = load_bfrag(wbuf, 5, nt, lane);

    // x2 = x*Wr1 + agg*Wl1 + b1
    v8f cx;
#pragma unroll
    for (int g = 0; g < 8; ++g) cx[g] = b1[col];
    cx = __builtin_amdgcn_wmma_f32_16x16x32_f16(false, a_x, false, br1, (short)0, cx, false, false);
    cx = __builtin_amdgcn_wmma_f32_16x16x32_f16(false, a_g, false, bl1, (short)0, cx, false, false);

    // y2 = x*Wr2 + agg*Wl2 + b2
    v8f cy;
#pragma unroll
    for (int g = 0; g < 8; ++g) cy[g] = b2[col];
    cy = __builtin_amdgcn_wmma_f32_16x16x32_f16(false, a_x, false, br2, (short)0, cy, false, false);
    cy = __builtin_amdgcn_wmma_f32_16x16x32_f16(false, a_g, false, bl2, (short)0, cy, false, false);

    // x3 = x*Wr3 + agg*Wl3 + b3  (stored; reused in gather kernel)
    v8f c3;
#pragma unroll
    for (int g = 0; g < 8; ++g) c3[g] = b3[col];
    c3 = __builtin_amdgcn_wmma_f32_16x16x32_f16(false, a_x, false, br3, (short)0, c3, false, false);
    c3 = __builtin_amdgcn_wmma_f32_16x16x32_f16(false, a_g, false, bl3, (short)0, c3, false, false);

    // C layout: VGPR g holds row (rowbase + g + 8*hs), column col.
#pragma unroll
    for (int g = 0; g < 8; ++g) {
      x3[(long long)(rowbase + g + 8 * hs) * DF + col] = c3[g];
      sg[g] += cx[g] * cy[g];
    }
  }

  // Row dot-product reduction: sum over the 16 lanes of each half.
#pragma unroll
  for (int g = 0; g < 8; ++g) {
    float s = sg[g];
    s += __shfl_xor(s, 8, 32);
    s += __shfl_xor(s, 4, 32);
    s += __shfl_xor(s, 2, 32);
    s += __shfl_xor(s, 1, 32);
    sg[g] = s;
  }
  if (m == 0) {
#pragma unroll
    for (int g = 0; g < 8; ++g)
      score[rowbase + 8 * hs + g] = sg[g] * 0.17677669529663689f;  // 1/sqrt(32)
  }
}

// ---------------------------------------------------------------- per-graph top-K
// Bitonic sort of 2048 (score,idx) pairs in LDS, descending; keep first 1024.
__global__ void __launch_bounds__(1024) topk_kernel(
    const float* __restrict__ score, int* __restrict__ perm, int* __restrict__ new_idx,
    float* __restrict__ out_nbatch, float* __restrict__ out_perm, float* __restrict__ out_score) {
  __shared__ float sv[NPG];
  __shared__ int   si[NPG];
  int g   = blockIdx.x;
  int tid = threadIdx.x;
  const float* sc = score + (long long)g * NPG;
  sv[tid] = sc[tid];               si[tid] = tid;
  sv[tid + 1024] = sc[tid + 1024]; si[tid + 1024] = tid + 1024;
  __syncthreads();

  for (int k = 2; k <= NPG; k <<= 1) {
    for (int j = k >> 1; j > 0; j >>= 1) {
      for (int t = tid; t < NPG; t += 1024) {
        int ixj = t ^ j;
        if (ixj > t) {
          bool dir = ((t & k) == 0);                 // true -> descending segment
          if ((sv[t] < sv[ixj]) == dir) {
            float tv = sv[t]; sv[t] = sv[ixj]; sv[ixj] = tv;
            int   ti = si[t]; si[t] = si[ixj]; si[ixj] = ti;
          }
        }
      }
      __syncthreads();
    }
  }

  if (tid < KKEEP) {
    int node = g * NPG + si[tid];
    int j    = g * KKEEP + tid;
    perm[j]       = node;
    new_idx[node] = j;
    out_perm[j]   = (float)node;
    out_score[j]  = sv[tid];
    out_nbatch[j] = (float)g;
  }
}

// ---------------------------------------------------------------- edge relabel
__global__ void edge_relabel(const int* __restrict__ src, const int* __restrict__ dst,
                             const float* __restrict__ eattr, const int* __restrict__ new_idx,
                             float* __restrict__ out_ei, float* __restrict__ out_ea) {
  int e = blockIdx.x * blockDim.x + threadIdx.x;
  if (e >= ETOT) return;
  int ns = new_idx[src[e]];
  int nd = new_idx[dst[e]];
  bool valid = (ns >= 0) && (nd >= 0);
  out_ei[e]        = (float)(valid ? ns : -1);
  out_ei[ETOT + e] = (float)(valid ? nd : -1);
  out_ea[e]        = valid ? eattr[e] : 0.0f;
}

// ---------------------------------------------------------------- x5 gather
__global__ void x5_kernel(const float* __restrict__ x, const float* __restrict__ x3,
                          const float* __restrict__ score, const int* __restrict__ perm,
                          float* __restrict__ out_x5) {
  int idx = blockIdx.x * blockDim.x + threadIdx.x;   // NKEEP * 8 threads
  int j  = idx >> 3;
  int d4 = idx & 7;
  if (j >= NKEEP) return;
  int p   = perm[j];
  float s = score[p];
  float4 a = ((const float4*)(x  + (long long)p * DF))[d4];
  float4 b = ((const float4*)(x3 + (long long)p * DF))[d4];
  float4 r = make_float4(a.x + s * b.x, a.y + s * b.y, a.z + s * b.z, a.w + s * b.w);
  ((float4*)(out_x5 + (long long)j * DF))[d4] = r;
}

// ================================================================ launch
extern "C" void kernel_launch(void* const* d_in, const int* in_sizes, int n_in,
                              void* d_out, int out_size, void* d_ws, size_t ws_size,
                              hipStream_t stream) {
  const float* x     = (const float*)d_in[0];
  const int*   src   = (const int*)d_in[1];          // edge_index row 0
  const int*   dst   = src + ETOT;                   // edge_index row 1
  const float* eattr = (const float*)d_in[2];
  // d_in[3] = batch (recomputed as p >> 11, equal-size graphs)
  const float* Wr1 = (const float*)d_in[4];
  const float* Wl1 = (const float*)d_in[5];
  const float* b1  = (const float*)d_in[6];
  const float* Wr2 = (const float*)d_in[7];
  const float* Wl2 = (const float*)d_in[8];
  const float* b2  = (const float*)d_in[9];
  const float* Wr3 = (const float*)d_in[10];
  const float* Wl3 = (const float*)d_in[11];
  const float* b3  = (const float*)d_in[12];

  // workspace layout
  float*     agg     = (float*)d_ws;                         // N*D
  float*     x3      = agg + (size_t)NTOT * DF;              // N*D
  float*     score   = x3 + (size_t)NTOT * DF;               // N
  int*       perm    = (int*)(score + NTOT);                 // NKEEP
  int*       new_idx = perm + NKEEP;                         // N
  _Float16*  wbuf    = (_Float16*)(new_idx + NTOT);          // 6*2*32*16 halves (12KB)

  // output layout (flat float, reference return order)
  float* out_x5     = (float*)d_out;                         // NKEEP*D
  float* out_ei     = out_x5 + (size_t)NKEEP * DF;           // 2*E
  float* out_ea     = out_ei + 2 * (size_t)ETOT;             // E
  float* out_nbatch = out_ea + (size_t)ETOT;                 // NKEEP
  float* out_perm   = out_nbatch + NKEEP;                    // NKEEP
  float* out_score  = out_perm + NKEEP;                      // NKEEP

  hipLaunchKernelGGL(fill_f32, dim3(((size_t)NTOT * DF + 255) / 256), dim3(256), 0, stream,
                     agg, 0.0f, (long long)NTOT * DF);
  hipLaunchKernelGGL(fill_i32, dim3((NTOT + 255) / 256), dim3(256), 0, stream,
                     new_idx, -1, NTOT);
  hipLaunchKernelGGL(prep_weights, dim3(1), dim3(512), 0, stream,
                     Wr1, Wl1, Wr2, Wl2, Wr3, Wl3, wbuf);
  hipLaunchKernelGGL(edge_scatter, dim3(((long long)ETOT * 32) / 256), dim3(256), 0, stream,
                     x, src, dst, agg);
  hipLaunchKernelGGL(gemm_score, dim3(NTOT / 128), dim3(256), 0, stream,
                     x, agg, wbuf, b1, b2, b3, x3, score);
  hipLaunchKernelGGL(topk_kernel, dim3(NB), dim3(1024), 0, stream,
                     score, perm, new_idx, out_nbatch, out_perm, out_score);
  hipLaunchKernelGGL(edge_relabel, dim3((ETOT + 255) / 256), dim3(256), 0, stream,
                     src, dst, eattr, new_idx, out_ei, out_ea);
  hipLaunchKernelGGL(x5_kernel, dim3((NKEEP * 8 + 255) / 256), dim3(256), 0, stream,
                     x, x3, score, perm, out_x5);
}
